// LSTM_16947940950306
// MI455X (gfx1250) — matrix operand order repaired and checked by reading, not compiled, not run on hardware
//
#include <hip/hip_runtime.h>
#include <hip/hip_bf16.h>

// ---------------------------------------------------------------------------
// LSTM(relu) for MI455X (gfx1250, wave32, WMMA).
// Fused recurrence: z_t = [x_t | h_t] @ [W;U] + b  (M=64, N=2048, K=640)
// Persistent kernel, 32 WGs x 128 thr; WG w owns 16 hidden units -> computes
// its 4 gate column-tiles; cell state c stays in VGPRs for all 1024 steps.
// Split-barrier pipelining: the x-part of step t+1 (K=0..127, independent of
// h) executes between barrier-signal and barrier-wait, hiding sync latency.
// ---------------------------------------------------------------------------

typedef __bf16 bf16;
typedef __attribute__((ext_vector_type(16))) __bf16 v16bf;
typedef __attribute__((ext_vector_type(8)))  __bf16 v8bf;
typedef __attribute__((ext_vector_type(8)))  float  v8f;

constexpr int kB = 64;                // batch
constexpr int kS = 1024;              // seq len
constexpr int kF = 128;               // features
constexpr int kH = 512;               // hidden
constexpr int kG = 4 * kH;            // 2048 gate columns
constexpr int kK = kF + kH;           // 640 = fused K
constexpr int kKB = kK / 32;          // 20 k-blocks of 32
constexpr int kKBx = kF / 32;         // 4 x-only k-blocks
constexpr int kNWG = kH / 16;         // 32 persistent workgroups
constexpr int kFragElems = 32 * 16;   // 512 bf16 per 32x16 B-fragment
constexpr size_t kHbufElems = (size_t)2 * kB * kH;          // double buffer
constexpr size_t kXsElems   = (size_t)kS * kB * kF;         // time-major x

__device__ __forceinline__ float sigmoidf_(float x) {
    return 1.0f / (1.0f + __expf(-x));
}

__device__ __forceinline__ v16bf load_frag16(const bf16* p0, const bf16* p1) {
    union { v16bf v; v8bf h[2]; } u;
    u.h[0] = *(const v8bf*)p0;
    u.h[1] = *(const v8bf*)p1;
    return u.v;
}

// ---- init: zero barrier counter + h double-buffer (every launch: replay-safe)
__global__ void lstm_init_ws(unsigned* __restrict__ bar, bf16* __restrict__ hbuf) {
    size_t i = (size_t)blockIdx.x * blockDim.x + threadIdx.x;
    if (i == 0) *bar = 0u;
    if (i < kHbufElems) hbuf[i] = (bf16)0.0f;
}

// ---- convert x [B,S,F] f32 -> xs [S,B,F] bf16 (time-major, WMMA-A friendly)
__global__ void lstm_convert_x(const float* __restrict__ x, bf16* __restrict__ xs) {
    size_t i = (size_t)blockIdx.x * blockDim.x + threadIdx.x;
    if (i >= kXsElems) return;
    int f = (int)(i % kF);
    size_t r = i / kF;
    int b = (int)(r % kB);
    int t = (int)(r / kB);
    xs[i] = (bf16)x[((size_t)b * kS + t) * kF + f];
}

// ---- pack [W;U] f32 -> bf16 B-fragments in this kernel's contiguous-k layout:
// wp[((wg*4+gate)*20+kb)*512 + lane*16 + e]; lane: N=lane&15, K=(lane>>4)*16+e
__global__ void lstm_pack_weights(const float* __restrict__ W,
                                  const float* __restrict__ U,
                                  bf16* __restrict__ wp) {
    int tid = blockIdx.x * blockDim.x + threadIdx.x;     // kNWG*4*kKB*32 threads
    if (tid >= kNWG * 4 * kKB * 32) return;
    int lane  = tid & 31;
    int rest  = tid >> 5;
    int kb    = rest % kKB;
    int rest2 = rest / kKB;
    int gate  = rest2 & 3;
    int wg    = rest2 >> 2;
    int col = gate * kH + wg * 16 + (lane & 15);
    int kbase = kb * 32 + (lane >> 4) * 16;
    bf16* dst = wp + (size_t)tid * 16;
#pragma unroll
    for (int e = 0; e < 16; ++e) {
        int K = kbase + e;
        float v = (K < kF) ? W[(size_t)K * kG + col]
                           : U[(size_t)(K - kF) * kG + col];
        dst[e] = (bf16)v;
    }
}

// ---- persistent recurrent kernel -------------------------------------------
__launch_bounds__(128, 1)
__global__ void lstm_persistent(const bf16* __restrict__ xs,
                                const bf16* __restrict__ wp,
                                const float* __restrict__ bias,
                                float* __restrict__ out,
                                bf16* __restrict__ hbuf,
                                unsigned* __restrict__ bar) {
    extern __shared__ bf16 ldsB[];                 // [4*kKB][32][16] = 80 KB
    const int wg   = blockIdx.x;                   // 0..31
    const int lane = threadIdx.x & 31;
    const int wave = threadIdx.x >> 5;             // 0..3 == M-tile (batch rows)

    // Stage this WG's pre-packed B fragments into LDS (one-time).
    const bf16* wpw = wp + (size_t)wg * (4 * kKB * kFragElems);
    for (int fidx = wave; fidx < 4 * kKB; fidx += 4) {
        const bf16* src = wpw + (size_t)fidx * kFragElems + lane * 16;
        bf16* dst = ldsB + (size_t)fidx * kFragElems + lane * 16;
        *(v8bf*)dst       = *(const v8bf*)src;
        *(v8bf*)(dst + 8) = *(const v8bf*)(src + 8);
    }
    __syncthreads();

    const int n16   = lane & 15;
    const int ubase = wg * 16 + n16;               // this lane's hidden unit
    float bi[4];
#pragma unroll
    for (int q = 0; q < 4; ++q) bi[q] = bias[q * kH + ubase];

    const int arow = wave * 16 + n16;              // A-matrix row (batch) for loads
    const int kb8  = 16 * (lane >> 4);             // A-layout K sub-offset
    const int bofs = wave * 16 + 8 * (lane >> 4);  // C/D-matrix row base

    v8f cacc;
#pragma unroll
    for (int e = 0; e < 8; ++e) cacc[e] = 0.0f;

    v8f acc0, acc1, acc2, acc3;

    // x-part for timestep t: init accumulators with bias and fold K=0..127.
    // Depends only on xs + LDS weights -> runs in the barrier shadow.
    auto xpart = [&](int t) {
#pragma unroll
        for (int e = 0; e < 8; ++e) {
            acc0[e] = bi[0]; acc1[e] = bi[1]; acc2[e] = bi[2]; acc3[e] = bi[3];
        }
        const bf16* xrow = xs + ((size_t)t * kB + arow) * kF;
#pragma unroll
        for (int kb = 0; kb < kKBx; ++kb) {
            const bf16* p = xrow + kb * 32 + kb8;
            v16bf a = load_frag16(p, p + 8);
            const bf16* b0 = ldsB + ((size_t)(0 * kKB + kb) * 32 + lane) * 16;
            const bf16* b1 = ldsB + ((size_t)(1 * kKB + kb) * 32 + lane) * 16;
            const bf16* b2 = ldsB + ((size_t)(2 * kKB + kb) * 32 + lane) * 16;
            const bf16* b3 = ldsB + ((size_t)(3 * kKB + kb) * 32 + lane) * 16;
            acc0 = __builtin_amdgcn_wmma_f32_16x16x32_bf16(
                false, a, false, load_frag16(b0, b0 + 8), (short)0, acc0, false, false);
            acc1 = __builtin_amdgcn_wmma_f32_16x16x32_bf16(
                false, a, false, load_frag16(b1, b1 + 8), (short)0, acc1, false, false);
            acc2 = __builtin_amdgcn_wmma_f32_16x16x32_bf16(
                false, a, false, load_frag16(b2, b2 + 8), (short)0, acc2, false, false);
            acc3 = __builtin_amdgcn_wmma_f32_16x16x32_bf16(
                false, a, false, load_frag16(b3, b3 + 8), (short)0, acc3, false, false);
        }
    };

    xpart(0);                                      // prologue (h0 == 0, no wait)

    for (int t = 0; t < kS; ++t) {
        // ---- wait for h_t to be globally visible (skip t=0: init-zeroed) ----
        if (t > 0) {
            __syncthreads();
            if (threadIdx.x == 0) {
                unsigned target = (unsigned)t * (unsigned)kNWG;
                while (atomicAdd(bar, 0u) < target) __builtin_amdgcn_s_sleep(8);
            }
            __syncthreads();
            __threadfence();                       // acquire
        }

        const bf16* hcur  = hbuf + (size_t)(t & 1) * kB * kH;
        bf16*       hnext = hbuf + (size_t)((t + 1) & 1) * kB * kH;
        const bf16* hrow  = hcur + (size_t)arow * kH;

        if (t + 1 < kS)
            __builtin_prefetch(xs + ((size_t)(t + 1) * kB + arow) * kF + kb8, 0, 1);

        // ---- h-dependent part: K = 128..639 (16 k-blocks) ----
#pragma unroll
        for (int kb = kKBx; kb < kKB; ++kb) {
            const bf16* p = hrow + (kb - kKBx) * 32 + kb8;
            v16bf a = load_frag16(p, p + 8);
            const bf16* b0 = ldsB + ((size_t)(0 * kKB + kb) * 32 + lane) * 16;
            const bf16* b1 = ldsB + ((size_t)(1 * kKB + kb) * 32 + lane) * 16;
            const bf16* b2 = ldsB + ((size_t)(2 * kKB + kb) * 32 + lane) * 16;
            const bf16* b3 = ldsB + ((size_t)(3 * kKB + kb) * 32 + lane) * 16;
            acc0 = __builtin_amdgcn_wmma_f32_16x16x32_bf16(
                false, a, false, load_frag16(b0, b0 + 8), (short)0, acc0, false, false);
            acc1 = __builtin_amdgcn_wmma_f32_16x16x32_bf16(
                false, a, false, load_frag16(b1, b1 + 8), (short)0, acc1, false, false);
            acc2 = __builtin_amdgcn_wmma_f32_16x16x32_bf16(
                false, a, false, load_frag16(b2, b2 + 8), (short)0, acc2, false, false);
            acc3 = __builtin_amdgcn_wmma_f32_16x16x32_bf16(
                false, a, false, load_frag16(b3, b3 + 8), (short)0, acc3, false, false);
        }

        // ---- gates (Keras order i,f,g,o; activation='relu') ----
#pragma unroll
        for (int e = 0; e < 8; ++e) {
            float ig = sigmoidf_(acc0[e]);
            float fg = sigmoidf_(acc1[e]);
            float g  = fmaxf(acc2[e], 0.0f);
            float og = sigmoidf_(acc3[e]);
            float cn = fg * cacc[e] + ig * g;
            cacc[e] = cn;
            float hv = og * fmaxf(cn, 0.0f);
            int brow = bofs + e;
            out[((size_t)brow * kS + t) * kH + ubase] = hv;
            hnext[(size_t)brow * kH + ubase] = (bf16)hv;
        }

        // ---- signal step t complete (release) ----
        __threadfence();
        __syncthreads();
        if (threadIdx.x == 0) atomicAdd(bar, 1u);

        // ---- barrier shadow: independent x-part of step t+1 ----
        if (t + 1 < kS) xpart(t + 1);
    }
}

// ---------------------------------------------------------------------------
extern "C" void kernel_launch(void* const* d_in, const int* in_sizes, int n_in,
                              void* d_out, int out_size, void* d_ws, size_t ws_size,
                              hipStream_t stream) {
    const float* x  = (const float*)d_in[0];   // [B,S,F]
    const float* W  = (const float*)d_in[1];   // [F,4H]
    const float* U  = (const float*)d_in[2];   // [H,4H]
    const float* bv = (const float*)d_in[3];   // [4H]
    float* out = (float*)d_out;                // [B,S,H]

    char* ws = (char*)d_ws;
    unsigned* bar = (unsigned*)ws;                                   // 256 B slot
    bf16* hbuf = (bf16*)(ws + 256);                                  // 128 KB
    bf16* xs   = (bf16*)(ws + 256 + kHbufElems * sizeof(bf16));      // 16 MB
    bf16* wp   = (bf16*)(ws + 256 + (kHbufElems + kXsElems) * sizeof(bf16)); // 2.6 MB

    // 1) reset barrier + zero h0 (replay-safe)
    {
        int total = (int)kHbufElems;
        lstm_init_ws<<<(total + 255) / 256, 256, 0, stream>>>(bar, hbuf);
    }
    // 2) x -> time-major bf16
    {
        size_t total = kXsElems;
        lstm_convert_x<<<(unsigned)((total + 255) / 256), 256, 0, stream>>>(x, xs);
    }
    // 3) pack [W;U] into WMMA B-fragment layout (bf16)
    {
        int total = kNWG * 4 * kKB * 32;
        lstm_pack_weights<<<(total + 255) / 256, 256, 0, stream>>>(W, U, wp);
    }
    // 4) persistent fused recurrence
    {
        size_t smem = (size_t)4 * kKB * kFragElems * sizeof(bf16);   // 80 KB
        lstm_persistent<<<kNWG, 128, smem, stream>>>(xs, wp, bv, out, hbuf, bar);
    }
}
